// Attention_52699248722576
// MI455X (gfx1250) — matrix operand order repaired and checked by reading, not compiled, not run on hardware
//
#include <hip/hip_runtime.h>
#include <hip/hip_bf16.h>
#include <stdint.h>

// ---------------- problem constants ----------------
#define DIMC   768
#define HEADS  12
#define HD     64          // head dim
#define BATCH  4
#define SEQ    2048
#define MTOT   (BATCH*SEQ) // 8192 tokens
#define FQKV   (3*DIMC)    // 2304

typedef __attribute__((ext_vector_type(16))) __bf16 v16bf;
typedef __attribute__((ext_vector_type(8)))  float  v8f;
typedef __attribute__((__vector_size__(4 * sizeof(int)))) int vi4;

union ABFrag {
  v16bf  v;
  uint4  u4[2];
  __bf16 h[16];
};
union CFrag {
  v8f   v;
  float f[8];
};

// ---------------- WMMA wrapper ----------------
__device__ inline v8f wmma_bf16(v16bf a, v16bf b, v8f c) {
  // (neg_a, A, neg_b, B, c_mod, C, reuse_a, reuse_b)
  return __builtin_amdgcn_wmma_f32_16x16x32_bf16(false, a, false, b, (short)0, c,
                                                 false, false);
}

// ---------------- async global->LDS staging (CDNA5 ASYNCcnt path) ----------------
#if defined(__has_builtin)
#if __has_builtin(__builtin_amdgcn_global_load_async_to_lds_b128)
#define HAVE_ASYNC_LDS 1
#endif
#endif

__device__ inline void g2s_b128(__bf16* lds, const __bf16* g) {
#ifdef HAVE_ASYNC_LDS
  __builtin_amdgcn_global_load_async_to_lds_b128(
      (vi4 __attribute__((address_space(1)))*)g,
      (vi4 __attribute__((address_space(3)))*)lds, 0, 0);
#else
  *reinterpret_cast<uint4*>(lds) = *reinterpret_cast<const uint4*>(g);
#endif
}

__device__ inline void wait_lds_stage() {
#ifdef HAVE_ASYNC_LDS
#if __has_builtin(__builtin_amdgcn_s_wait_asynccnt)
  __builtin_amdgcn_s_wait_asynccnt(0);
#else
  asm volatile("s_wait_asynccnt 0" ::: "memory");
#endif
#endif
}

// ---------------- fragment loaders (wave32 layouts per ISA 7.12.2) ----------------
// A-matrix 16x32 bf16, source row-major [16 x ld], tile origin `base`, K offset k0.
// lane L: row = L&15 ; kh = L>>4 ; VGPR0-3 <- K = kh*8+0..7 ; VGPR4-7 <- K = 16+kh*8+0..7
__device__ inline v16bf load_a_rowmajor(const __bf16* base, int ld, int k0, int lane) {
  int row = lane & 15;
  int kh  = (lane >> 4) & 1;
  const __bf16* p = base + (size_t)row * ld + k0 + kh * 8;
  ABFrag a;
  a.u4[0] = *reinterpret_cast<const uint4*>(p);
  a.u4[1] = *reinterpret_cast<const uint4*>(p + 16);
  return a.v;
}

// B-matrix 32x16 bf16 for C = A * S^T where S is row-major [n x ld] (rows contiguous in K).
// lane L: col n = L&15 ; kh = L>>4 ; VGPR0-3 <- K = kh*8+0..7 ; VGPR4-7 <- K = 16+kh*8+0..7
__device__ inline v16bf load_bT(const __bf16* base, int ld, int k0, int lane) {
  int n  = lane & 15;
  int kh = (lane >> 4) & 1;
  const __bf16* p = base + (size_t)n * ld + k0 + kh * 8;
  ABFrag b;
  b.u4[0] = *reinterpret_cast<const uint4*>(p);
  b.u4[1] = *reinterpret_cast<const uint4*>(p + 16);
  return b.v;
}

// ---------------- kernel 0: fp32 -> bf16 convert ----------------
__global__ void cvt_f32_bf16(const float* __restrict__ src, __bf16* __restrict__ dst, int n) {
  int i = blockIdx.x * blockDim.x + threadIdx.x;
  if (i < n) dst[i] = (__bf16)src[i];
}

// ---------------- kernel 1: LayerNorm + bf16 cast ----------------
__global__ void __launch_bounds__(256) layernorm_bf16(const float* __restrict__ x,
                                                      const float* __restrict__ w,
                                                      const float* __restrict__ bias,
                                                      __bf16* __restrict__ xn) {
  __shared__ float red[256];
  int row = blockIdx.x;
  int t   = threadIdx.x;
  const float* xr = x + (size_t)row * DIMC;
  float v0 = xr[t], v1 = xr[t + 256], v2 = xr[t + 512];
  red[t] = v0 + v1 + v2;
  __syncthreads();
#pragma unroll
  for (int off = 128; off > 0; off >>= 1) {
    if (t < off) red[t] += red[t + off];
    __syncthreads();
  }
  float mu = red[0] * (1.0f / DIMC);
  __syncthreads();
  float d0 = v0 - mu, d1 = v1 - mu, d2 = v2 - mu;
  red[t] = d0 * d0 + d1 * d1 + d2 * d2;
  __syncthreads();
#pragma unroll
  for (int off = 128; off > 0; off >>= 1) {
    if (t < off) red[t] += red[t + off];
    __syncthreads();
  }
  float rstd = rsqrtf(red[0] * (1.0f / DIMC) + 1e-5f);
  __bf16* o = xn + (size_t)row * DIMC;
  o[t]       = (__bf16)(d0 * rstd * w[t]       + bias[t]);
  o[t + 256] = (__bf16)(d1 * rstd * w[t + 256] + bias[t + 256]);
  o[t + 512] = (__bf16)(d2 * rstd * w[t + 512] + bias[t + 512]);
}

// ---------------- kernel 2: QKV GEMM (32x64 per wave), scatter heads ----------------
// Q,K stored [b,h,n,64] ; V stored TRANSPOSED [b,h,64,n] so attention's P*V
// B-fragments become contiguous 16B loads.
__global__ void __launch_bounds__(256) qkv_gemm(const __bf16* __restrict__ xn,
                                                const __bf16* __restrict__ wq,
                                                __bf16* __restrict__ qb,
                                                __bf16* __restrict__ kb,
                                                __bf16* __restrict__ vt) {
  int lane = threadIdx.x & 31;
  int wid  = blockIdx.x * (blockDim.x >> 5) + (threadIdx.x >> 5);
  const int TM = MTOT / 32;        // 256 row tiles of 32
  int tile_m = wid % TM;
  int tile_f = wid / TM;           // 0..35, each 64 output cols == one head of q/k/v
  const __bf16* A = xn + (size_t)tile_m * 32 * DIMC;
  const __bf16* W = wq + (size_t)tile_f * 64 * DIMC;

  v8f acc0[4] = {}, acc1[4] = {};
  for (int k0 = 0; k0 < DIMC; k0 += 32) {
    __builtin_prefetch(A + (size_t)(lane & 15) * DIMC + k0 + 64, 0, 3);
    v16bf a0 = load_a_rowmajor(A, DIMC, k0, lane);
    v16bf a1 = load_a_rowmajor(A + (size_t)16 * DIMC, DIMC, k0, lane);
#pragma unroll
    for (int s = 0; s < 4; ++s) {
      v16bf b = load_bT(W + (size_t)s * 16 * DIMC, DIMC, k0, lane);
      acc0[s] = wmma_bf16(a0, b, acc0[s]);
      acc1[s] = wmma_bf16(a1, b, acc1[s]);
    }
  }

  int which = tile_f / HEADS;      // 0=q 1=k 2=v
  int h     = tile_f % HEADS;
  int kh = (lane >> 4) & 1;
  int cl = lane & 15;
#pragma unroll
  for (int half = 0; half < 2; ++half) {
#pragma unroll
    for (int s = 0; s < 4; ++s) {
      CFrag c; c.v = half ? acc1[s] : acc0[s];
#pragma unroll
      for (int r = 0; r < 8; ++r) {
        int row = tile_m * 32 + half * 16 + r + 8 * kh;  // token index
        int bb  = row >> 11;
        int n   = row & (SEQ - 1);
        int dd  = s * 16 + cl;
        __bf16 val = (__bf16)c.f[r];
        if (which < 2) {
          __bf16* dst = (which == 0) ? qb : kb;
          dst[(((size_t)bb * HEADS + h) * SEQ + n) * HD + dd] = val;
        } else {
          vt[(((size_t)bb * HEADS + h) * HD + dd) * SEQ + n] = val;
        }
      }
    }
  }
}

// ---------------- kernel 3: flash attention (cooperative K/V LDS staging) ----------------
// block: 8 waves = 128 q-rows of one (b,h). All waves sweep keys in lockstep chunks
// of 32; K chunk (32x64) and V^T chunk (64x32) are staged into LDS once per block.
__global__ void __launch_bounds__(256) attn_kernel(const __bf16* __restrict__ qb,
                                                   const __bf16* __restrict__ kb,
                                                   const __bf16* __restrict__ vt,
                                                   __bf16* __restrict__ ob) {
  __shared__ __align__(16) __bf16 kbufS[32 * 64];     // K  rows j0..j0+31, row-major
  __shared__ __align__(16) __bf16 vbufS[64 * 32];     // V^T rows dd=0..63, cols j0..j0+31
  __shared__ __align__(16) __bf16 pbuf[8][16 * 32];   // per-wave P transpose staging
  int t    = threadIdx.x;
  int lane = t & 31;
  int wv   = t >> 5;
  int bh   = blockIdx.x / (SEQ / 128);
  int mblk = blockIdx.x % (SEQ / 128);
  int qrow0 = mblk * 128 + wv * 16;

  const __bf16* Q  = qb + (size_t)bh * SEQ * HD + (size_t)qrow0 * HD;
  const __bf16* K  = kb + (size_t)bh * SEQ * HD;
  const __bf16* VT = vt + (size_t)bh * SEQ * HD;      // [64][SEQ]

  v16bf qa0 = load_a_rowmajor(Q, HD, 0, lane);
  v16bf qa1 = load_a_rowmajor(Q, HD, 32, lane);

  v8f   o[4] = {};
  float mrun[8], lsum[8];
#pragma unroll
  for (int r = 0; r < 8; ++r) { mrun[r] = -1e30f; lsum[r] = 0.0f; }
  const float scale = 0.125f;   // 1/sqrt(64)

  __bf16* plds = &pbuf[wv][0];
  int kh = (lane >> 4) & 1;
  int cl = lane & 15;

  // per-thread staging offsets: 4KB buffer = 256 threads x 16B
  int koff = (t >> 3) * 64 + (t & 7) * 8;             // K: row t/8, 8 elems
  int vrow = t >> 2, vcol8 = (t & 3) * 8;             // V^T: row t/4, 8 elems

  for (int j0 = 0; j0 < SEQ; j0 += 32) {
    // ---- cooperative stage: K(32x64) and V^T(64x32) chunks ----
    g2s_b128(&kbufS[koff], K + (size_t)j0 * HD + koff);
    g2s_b128(&vbufS[vrow * 32 + vcol8], VT + (size_t)vrow * SEQ + j0 + vcol8);
    wait_lds_stage();
    __syncthreads();

    // ---- S = Q * K^T for 32 keys (two 16-col C tiles) ----
    v8f s0 = {}, s1 = {};
    {
      v16bf b00 = load_bT(kbufS, HD, 0,  lane);
      v16bf b01 = load_bT(kbufS, HD, 32, lane);
      s0 = wmma_bf16(qa0, b00, s0);
      s0 = wmma_bf16(qa1, b01, s0);
      v16bf b10 = load_bT(kbufS + 16 * HD, HD, 0,  lane);
      v16bf b11 = load_bT(kbufS + 16 * HD, HD, 32, lane);
      s1 = wmma_bf16(qa0, b10, s1);
      s1 = wmma_bf16(qa1, b11, s1);
    }
    CFrag c0, c1; c0.v = s0; c1.v = s1;

    // ---- online softmax (row stats live inside each 16-lane half) ----
    float pl0[8], pl1[8], alpha[8];
#pragma unroll
    for (int r = 0; r < 8; ++r) {
      float a0 = c0.f[r] * scale;
      float a1 = c1.f[r] * scale;
      float mx = fmaxf(a0, a1);
      mx = fmaxf(mx, __shfl_xor(mx, 1, 32));
      mx = fmaxf(mx, __shfl_xor(mx, 2, 32));
      mx = fmaxf(mx, __shfl_xor(mx, 4, 32));
      mx = fmaxf(mx, __shfl_xor(mx, 8, 32));
      float mnew = fmaxf(mrun[r], mx);
      alpha[r] = __expf(mrun[r] - mnew);
      pl0[r]   = __expf(a0 - mnew);
      pl1[r]   = __expf(a1 - mnew);
      float rs = pl0[r] + pl1[r];
      rs += __shfl_xor(rs, 1, 32);
      rs += __shfl_xor(rs, 2, 32);
      rs += __shfl_xor(rs, 4, 32);
      rs += __shfl_xor(rs, 8, 32);
      lsum[r] = lsum[r] * alpha[r] + rs;
      mrun[r] = mnew;
    }
#pragma unroll
    for (int s = 0; s < 4; ++s) {
      CFrag tt; tt.v = o[s];
#pragma unroll
      for (int r = 0; r < 8; ++r) tt.f[r] *= alpha[r];
      o[s] = tt.v;
    }
    // ---- P: C-layout -> LDS (row-major 16x32) -> A-layout ----
#pragma unroll
    for (int r = 0; r < 8; ++r) {
      int m = r + 8 * kh;
      plds[m * 32 + cl]      = (__bf16)pl0[r];
      plds[m * 32 + 16 + cl] = (__bf16)pl1[r];
    }
    v16bf pa = load_a_rowmajor(plds, 32, 0, lane);

    // ---- O += P * V  (V^T rows contiguous in K -> plain bT loads from LDS) ----
#pragma unroll
    for (int s = 0; s < 4; ++s) {
      v16bf vf = load_bT(vbufS + (size_t)s * 16 * 32, 32, 0, lane);
      o[s] = wmma_bf16(pa, vf, o[s]);
    }
    __syncthreads();   // protect K/V LDS buffers before next stage
  }

  // ---- normalize and store to [b, n, h*64+dd] (bf16) ----
  int bb = bh / HEADS, hh = bh % HEADS;
#pragma unroll
  for (int s = 0; s < 4; ++s) {
    CFrag c; c.v = o[s];
#pragma unroll
    for (int r = 0; r < 8; ++r) {
      int n  = qrow0 + r + 8 * kh;
      int dd = s * 16 + cl;
      ob[((size_t)bb * SEQ + n) * DIMC + hh * HD + dd] = (__bf16)(c.f[r] / lsum[r]);
    }
  }
}

// ---------------- kernel 4: output projection + bias (fp32 out), 32x64/wave ----------------
__global__ void __launch_bounds__(256) out_gemm(const __bf16* __restrict__ attn,
                                                const __bf16* __restrict__ wo,
                                                const float* __restrict__ bo,
                                                float* __restrict__ out) {
  int lane = threadIdx.x & 31;
  int wid  = blockIdx.x * (blockDim.x >> 5) + (threadIdx.x >> 5);
  const int TM = MTOT / 32;       // 256
  int tile_m = wid % TM;
  int tile_f = wid / TM;          // 0..11
  const __bf16* A = attn + (size_t)tile_m * 32 * DIMC;
  const __bf16* W = wo + (size_t)tile_f * 64 * DIMC;

  v8f acc0[4] = {}, acc1[4] = {};
  for (int k0 = 0; k0 < DIMC; k0 += 32) {
    v16bf a0 = load_a_rowmajor(A, DIMC, k0, lane);
    v16bf a1 = load_a_rowmajor(A + (size_t)16 * DIMC, DIMC, k0, lane);
#pragma unroll
    for (int s = 0; s < 4; ++s) {
      v16bf b = load_bT(W + (size_t)s * 16 * DIMC, DIMC, k0, lane);
      acc0[s] = wmma_bf16(a0, b, acc0[s]);
      acc1[s] = wmma_bf16(a1, b, acc1[s]);
    }
  }
  int kh = (lane >> 4) & 1;
  int cl = lane & 15;
#pragma unroll
  for (int half = 0; half < 2; ++half) {
#pragma unroll
    for (int s = 0; s < 4; ++s) {
      CFrag c; c.v = half ? acc1[s] : acc0[s];
#pragma unroll
      for (int r = 0; r < 8; ++r) {
        int row = tile_m * 32 + half * 16 + r + 8 * kh;
        int col = tile_f * 64 + s * 16 + cl;
        out[(size_t)row * DIMC + col] = c.f[r] + bo[col];
      }
    }
  }
}

// ---------------- launcher ----------------
extern "C" void kernel_launch(void* const* d_in, const int* in_sizes, int n_in,
                              void* d_out, int out_size, void* d_ws, size_t ws_size,
                              hipStream_t stream) {
  (void)in_sizes; (void)n_in; (void)out_size; (void)ws_size;
  const float* x    = (const float*)d_in[0];
  const float* ln_w = (const float*)d_in[1];
  const float* ln_b = (const float*)d_in[2];
  const float* wqkv = (const float*)d_in[3];
  const float* wout = (const float*)d_in[4];
  const float* bout = (const float*)d_in[5];
  float* out = (float*)d_out;

  char*  ws  = (char*)d_ws;
  size_t off = 0;
  auto take = [&](size_t bytes) {
    void* p = ws + off;
    off += (bytes + 255) & ~(size_t)255;
    return p;
  };
  __bf16* xn  = (__bf16*)take((size_t)MTOT * DIMC * 2);
  __bf16* wqb = (__bf16*)take((size_t)FQKV * DIMC * 2);
  __bf16* wob = (__bf16*)take((size_t)DIMC * DIMC * 2);
  __bf16* qb  = (__bf16*)take((size_t)MTOT * DIMC * 2);   // [b,h,n,64]
  __bf16* kbf = (__bf16*)take((size_t)MTOT * DIMC * 2);   // [b,h,n,64]
  __bf16* vtb = (__bf16*)take((size_t)MTOT * DIMC * 2);   // [b,h,64,n]  (transposed)
  __bf16* ab  = (__bf16*)take((size_t)MTOT * DIMC * 2);   // attn out [b,n,768]

  {
    int n1 = FQKV * DIMC;
    cvt_f32_bf16<<<(n1 + 255) / 256, 256, 0, stream>>>(wqkv, wqb, n1);
    int n2 = DIMC * DIMC;
    cvt_f32_bf16<<<(n2 + 255) / 256, 256, 0, stream>>>(wout, wob, n2);
  }
  layernorm_bf16<<<MTOT, 256, 0, stream>>>(x, ln_w, ln_b, xn);
  // 256*36 waves / 8 per block
  qkv_gemm<<<(MTOT / 32) * (FQKV / 64) / 8, 256, 0, stream>>>(xn, wqb, qb, kbf, vtb);
  attn_kernel<<<BATCH * HEADS * (SEQ / 128), 256, 0, stream>>>(qb, kbf, vtb, ab);
  // 256*12 waves / 8 per block
  out_gemm<<<(MTOT / 32) * (DIMC / 64) / 8, 256, 0, stream>>>(ab, wob, bout, out);
}
